// GlobalProtoEMA_45518063403695
// MI455X (gfx1250) — compile-verified, matrix operand order checked by reading, not run on hardware
//
#include <hip/hip_runtime.h>

typedef __attribute__((ext_vector_type(16))) __bf16 v16bf;
typedef __attribute__((ext_vector_type(8)))  float  v8f;

#define TAU_INV (1.0f / 0.07f)

// K mapping for 16-bit A/B fragments (CDNA5 ISA 16-bit A-matrix 16x32 table):
// lanes<16 cover K {0..7,16..23}; lanes>=16 the same +8.
__device__ __forceinline__ int kmap16(int i, int koff) {
    int v = i >> 1, j = i & 1;
    int base = (v < 4) ? (2 * v) : (2 * v + 8);
    return base + koff + j;
}

// Inverse map for even k in [0,32): which lane-half (hi) and half-index (i0, even).
__device__ __forceinline__ void kinv(int k, int& hi, int& i0) {
    hi = (k >> 3) & 1;
    int v = ((k & 7) >> 1) + ((k >> 4) << 2);
    i0 = 2 * v;
}

__device__ __forceinline__ unsigned int pack_bf16(float lo, float hi) {
    unsigned short ls = __builtin_bit_cast(unsigned short, (__bf16)lo);
    unsigned short hs = __builtin_bit_cast(unsigned short, (__bf16)hi);
    return (unsigned int)ls | ((unsigned int)hs << 16);
}

__device__ __forceinline__ float half_shfl_sum(float v) {
    v += __shfl_xor(v, 1, 32);
    v += __shfl_xor(v, 2, 32);
    v += __shfl_xor(v, 4, 32);
    v += __shfl_xor(v, 8, 32);
    return v;
}

__device__ __forceinline__ int clamp31(int v) { return v < 0 ? 0 : (v > 31 ? 31 : v); }

// bilinear sample of a 32x32 plane at (py,px) of the 64x64 output grid,
// half-pixel centers (align_corners=False): src = 0.5*dst - 0.25
__device__ __forceinline__ float bilin32(const float* __restrict__ plane, int py, int px) {
    float sy = py * 0.5f - 0.25f;
    float fy = floorf(sy);
    float wy = sy - fy;
    int iy0 = clamp31((int)fy), iy1 = clamp31((int)fy + 1);
    float sx = px * 0.5f - 0.25f;
    float fx = floorf(sx);
    float wx = sx - fx;
    int ix0 = clamp31((int)fx), ix1 = clamp31((int)fx + 1);
    float a0 = plane[iy0 * 32 + ix0], a1 = plane[iy0 * 32 + ix1];
    float b0 = plane[iy1 * 32 + ix0], b1 = plane[iy1 * 32 + ix1];
    return (1.0f - wy) * ((1.0f - wx) * a0 + wx * a1) + wy * ((1.0f - wx) * b0 + wx * b1);
}

// Shared-memory partition offsets (bytes, all 32B aligned)
#define OFF_XF   0          // 8192: X fragments [p_tile*32+lane][16]  (teacher: reused as sN)
#define OFF_WF   8192       // 8192: W fragments [t*32+lane][16]
#define OFF_CF   16384      // 4096: codebook A fragments [kb*32+lane][16]
#define OFF_ZF   20480      // 32768: per-wave zs B fragments [wave][(kb*32+lane)*16]
#define OFF_H_S  53248      // student: sH (64)
#define SMEM_S   53312
#define OFF_QF   53248      // teacher: per-wave qt A fragments [wave][lane*16] (8192)
#define OFF_H_T  61440      // teacher: sH (64)
#define SMEM_T   61504

// ---------------------------------------------------------------------------
// Student: fs4 [B,256,64,64] x Ws [128,256] -> zs -> softmax(zs.C/tau) -> hs sums
// ---------------------------------------------------------------------------
__global__ __launch_bounds__(256)
void student_kernel(const float* __restrict__ fs4, const float* __restrict__ Ws,
                    const float* __restrict__ Cg, float* __restrict__ gHs)
{
    extern __shared__ __align__(32) char smem[];
    __bf16* sXf = (__bf16*)(smem + OFF_XF);
    __bf16* sWf = (__bf16*)(smem + OFF_WF);
    __bf16* sCf = (__bf16*)(smem + OFF_CF);
    float*  sH  = (float*)(smem + OFF_H_S);

    const int tid  = threadIdx.x;
    const int lane = tid & 31;
    const int wave = tid >> 5;
    const int col  = lane & 15;
    const int hi   = lane >> 4;
    const int koff = hi ? 8 : 0;

    const int gp0 = blockIdx.x * 128;
    const int b   = gp0 >> 12;
    const int pb  = gp0 & 4095;

    if (tid < 16) sH[tid] = 0.0f;
    __bf16* myZ = (__bf16*)(smem + OFF_ZF) + wave * 2048;

    // build codebook fragments once per block: sCf[(kb*32+laneF)*16 + i]
    #pragma unroll
    for (int j = 0; j < 8; ++j) {
        int h = tid * 8 + j;                 // 0..2047
        int i = h & 15, laneF = (h >> 4) & 31, kb = h >> 9;
        int m = laneF & 15, koffF = ((laneF >> 4) & 1) * 8;
        sCf[h] = (__bf16)Cg[m * 128 + kb * 32 + kmap16(i, koffF)];
    }

    v8f acc[8] = {};

    for (int kb = 0; kb < 8; ++kb) {
        __syncthreads();
        // stage X fragments: pairs of channels packed, coalesced b128 reads
        const size_t base_x = (((size_t)(b * 256 + kb * 32)) << 12) + pb;
        #pragma unroll
        for (int j = 0; j < 2; ++j) {
            int T = tid * 2 + j;             // 512 tiles: 16 cpair x 32 p4
            int c0 = (T >> 5) * 2;
            int p4 = (T & 31) << 2;
            const float4 va = *(const float4*)(fs4 + base_x + (size_t)(c0 + 0) * 4096 + p4);
            const float4 vb = *(const float4*)(fs4 + base_x + (size_t)(c0 + 1) * 4096 + p4);
            int hiL, i0; kinv(c0, hiL, i0);
            const float* pa = (const float*)&va;
            const float* pbv = (const float*)&vb;
            #pragma unroll
            for (int pp = 0; pp < 4; ++pp) {
                int pl = p4 + pp;
                int chunk = (pl >> 4) * 32 + (pl & 15) + 16 * hiL;
                ((unsigned int*)sXf)[chunk * 8 + (i0 >> 1)] = pack_bf16(pa[pp], pbv[pp]);
            }
        }
        // stage W fragments: Ws[dd, kb*32 + c]
        #pragma unroll
        for (int j4 = 0; j4 < 4; ++j4) {
            int idx = tid * 16 + j4 * 4;
            int dd = idx >> 5, c0 = idx & 31;
            const float4 v = *(const float4*)(Ws + (size_t)dd * 256 + kb * 32 + c0);
            int t = dd >> 4, lane_lo = dd & 15;
            int hiA, iA; kinv(c0, hiA, iA);
            ((unsigned int*)sWf)[(t * 32 + lane_lo + 16 * hiA) * 8 + (iA >> 1)] = pack_bf16(v.x, v.y);
            int hiB, iB; kinv(c0 + 2, hiB, iB);
            ((unsigned int*)sWf)[(t * 32 + lane_lo + 16 * hiB) * 8 + (iB >> 1)] = pack_bf16(v.z, v.w);
        }
        __syncthreads();

        const v16bf a = *((const v16bf*)sXf + (wave * 32 + lane));
        #pragma unroll
        for (int t = 0; t < 8; ++t) {
            const v16bf bfr = *((const v16bf*)sWf + (t * 32 + lane));
            acc[t] = __builtin_amdgcn_wmma_f32_16x16x32_bf16(false, a, false, bfr,
                                                             (short)0, acc[t], false, false);
        }
    }

    // per-pixel l2 norm (pixel = r + 8*hi, spread across lanes of the half)
    float scale[8];
    #pragma unroll
    for (int r = 0; r < 8; ++r) {
        float ns = 0.0f;
        #pragma unroll
        for (int t = 0; t < 8; ++t) ns += acc[t][r] * acc[t][r];
        ns = half_shfl_sum(ns);
        scale[r] = 1.0f / fmaxf(sqrtf(ns), 1e-12f);
    }
    // scale + scatter zs into logits-B fragment layout
    #pragma unroll
    for (int t = 0; t < 8; ++t) {
        int dd = t * 16 + col;
        int ddl = dd & 31;
        int hiF = (ddl >> 3) & 1;
        int vF  = ((ddl & 7) >> 1) + ((ddl >> 4) << 2);
        int iF  = 2 * vF + (dd & 1);
        int kbF = dd >> 5;
        #pragma unroll
        for (int r = 0; r < 8; ++r) {
            float z = acc[t][r] * scale[r];
            acc[t][r] = z;
            myZ[(kbF * 32 + (r + 8 * hi) + 16 * hiF) * 16 + iF] = (__bf16)z;
        }
    }
    __syncthreads();

    // logits [m,p] = C (16x128) x zs^T (128x16): 4 bf16 WMMAs
    v8f lg = {};
    #pragma unroll
    for (int kb = 0; kb < 4; ++kb) {
        const v16bf a   = *((const v16bf*)sCf + (kb * 32 + lane));
        const v16bf bfr = *((const v16bf*)myZ + (kb * 32 + lane));
        lg = __builtin_amdgcn_wmma_f32_16x16x32_bf16(false, a, false, bfr,
                                                     (short)0, lg, false, false);
    }

    // softmax over m (16) per pixel column
    float mx = -3.0e38f;
    #pragma unroll
    for (int r = 0; r < 8; ++r) mx = fmaxf(mx, lg[r]);
    mx = fmaxf(mx, __shfl_xor(mx, 16, 32));
    float e[8], sm = 0.0f;
    #pragma unroll
    for (int r = 0; r < 8; ++r) { e[r] = __expf((lg[r] - mx) * TAU_INV); sm += e[r]; }
    sm += __shfl_xor(sm, 16, 32);
    float inv = 1.0f / sm;
    #pragma unroll
    for (int r = 0; r < 8; ++r) {
        float s = half_shfl_sum(e[r] * inv);
        if (col == 0) atomicAdd(&sH[r + 8 * hi], s);
    }
    __syncthreads();
    if (tid < 16) atomicAdd(&gHs[b * 16 + tid], sH[tid]);
}

// ---------------------------------------------------------------------------
// Teacher: bilinear-up ft [B,512,32,32] x Wt [128,512] -> zt -> qt -> ht sums
// plus mu numerator N = qt^T zt (bf16 WMMA, K zero-padded to 32).
// ---------------------------------------------------------------------------
__global__ __launch_bounds__(256)
void teacher_kernel(const float* __restrict__ ft, const float* __restrict__ Wt,
                    const float* __restrict__ Cg, float* __restrict__ gHt,
                    float* __restrict__ gN)
{
    extern __shared__ __align__(32) char smem[];
    __bf16* sXf = (__bf16*)(smem + OFF_XF);   // later reused as sN (float[2048])
    __bf16* sWf = (__bf16*)(smem + OFF_WF);
    __bf16* sCf = (__bf16*)(smem + OFF_CF);
    float*  sH  = (float*)(smem + OFF_H_T);

    const int tid  = threadIdx.x;
    const int lane = tid & 31;
    const int wave = tid >> 5;
    const int col  = lane & 15;
    const int hi   = lane >> 4;

    const int gp0 = blockIdx.x * 128;
    const int b   = gp0 >> 12;
    const int pb  = gp0 & 4095;
    const int py0 = pb >> 6;

    if (tid < 16) sH[tid] = 0.0f;
    __bf16* myZ = (__bf16*)(smem + OFF_ZF) + wave * 2048;
    __bf16* myQ = (__bf16*)(smem + OFF_QF) + wave * 512;

    #pragma unroll
    for (int j = 0; j < 8; ++j) {
        int h = tid * 8 + j;
        int i = h & 15, laneF = (h >> 4) & 31, kb = h >> 9;
        int m = laneF & 15, koffF = ((laneF >> 4) & 1) * 8;
        sCf[h] = (__bf16)Cg[m * 128 + kb * 32 + kmap16(i, koffF)];
    }

    v8f acc[8] = {};

    for (int kb = 0; kb < 16; ++kb) {
        __syncthreads();
        // stage upsampled X fragments (channel pairs packed)
        #pragma unroll
        for (int j = 0; j < 2; ++j) {
            int T = tid * 2 + j;
            int c0 = (T >> 5) * 2;
            int p4 = (T & 31) << 2;
            int hiL, i0; kinv(c0, hiL, i0);
            const float* pl0 = ft + (((size_t)(b * 512 + kb * 32 + c0 + 0)) << 10);
            const float* pl1 = ft + (((size_t)(b * 512 + kb * 32 + c0 + 1)) << 10);
            #pragma unroll
            for (int pp = 0; pp < 4; ++pp) {
                int pl = p4 + pp;
                int py = py0 + (pl >> 6), px = pl & 63;
                float lo = bilin32(pl0, py, px);
                float hv = bilin32(pl1, py, px);
                int chunk = (pl >> 4) * 32 + (pl & 15) + 16 * hiL;
                ((unsigned int*)sXf)[chunk * 8 + (i0 >> 1)] = pack_bf16(lo, hv);
            }
        }
        // stage W fragments
        #pragma unroll
        for (int j4 = 0; j4 < 4; ++j4) {
            int idx = tid * 16 + j4 * 4;
            int dd = idx >> 5, c0 = idx & 31;
            const float4 v = *(const float4*)(Wt + (size_t)dd * 512 + kb * 32 + c0);
            int t = dd >> 4, lane_lo = dd & 15;
            int hiA, iA; kinv(c0, hiA, iA);
            ((unsigned int*)sWf)[(t * 32 + lane_lo + 16 * hiA) * 8 + (iA >> 1)] = pack_bf16(v.x, v.y);
            int hiB, iB; kinv(c0 + 2, hiB, iB);
            ((unsigned int*)sWf)[(t * 32 + lane_lo + 16 * hiB) * 8 + (iB >> 1)] = pack_bf16(v.z, v.w);
        }
        __syncthreads();

        const v16bf a = *((const v16bf*)sXf + (wave * 32 + lane));
        #pragma unroll
        for (int t = 0; t < 8; ++t) {
            const v16bf bfr = *((const v16bf*)sWf + (t * 32 + lane));
            acc[t] = __builtin_amdgcn_wmma_f32_16x16x32_bf16(false, a, false, bfr,
                                                             (short)0, acc[t], false, false);
        }
    }
    __syncthreads();               // sXf region now free -> becomes sN
    float* sN = (float*)(smem + OFF_XF);
    for (int i2 = tid; i2 < 2048; i2 += 256) sN[i2] = 0.0f;

    // l2 norm + stage zt fragments
    float scale[8];
    #pragma unroll
    for (int r = 0; r < 8; ++r) {
        float ns = 0.0f;
        #pragma unroll
        for (int t = 0; t < 8; ++t) ns += acc[t][r] * acc[t][r];
        ns = half_shfl_sum(ns);
        scale[r] = 1.0f / fmaxf(sqrtf(ns), 1e-12f);
    }
    #pragma unroll
    for (int t = 0; t < 8; ++t) {
        int dd = t * 16 + col;
        int ddl = dd & 31;
        int hiF = (ddl >> 3) & 1;
        int vF  = ((ddl & 7) >> 1) + ((ddl >> 4) << 2);
        int iF  = 2 * vF + (dd & 1);
        int kbF = dd >> 5;
        #pragma unroll
        for (int r = 0; r < 8; ++r) {
            float z = acc[t][r] * scale[r];
            acc[t][r] = z;                         // keep zt in regs for mu-B reuse
            myZ[(kbF * 32 + (r + 8 * hi) + 16 * hiF) * 16 + iF] = (__bf16)z;
        }
    }
    __syncthreads();

    // logits
    v8f lg = {};
    #pragma unroll
    for (int kb = 0; kb < 4; ++kb) {
        const v16bf a   = *((const v16bf*)sCf + (kb * 32 + lane));
        const v16bf bfr = *((const v16bf*)myZ + (kb * 32 + lane));
        lg = __builtin_amdgcn_wmma_f32_16x16x32_bf16(false, a, false, bfr,
                                                     (short)0, lg, false, false);
    }

    // softmax over m
    float mx = -3.0e38f;
    #pragma unroll
    for (int r = 0; r < 8; ++r) mx = fmaxf(mx, lg[r]);
    mx = fmaxf(mx, __shfl_xor(mx, 16, 32));
    float e[8], sm = 0.0f;
    #pragma unroll
    for (int r = 0; r < 8; ++r) { e[r] = __expf((lg[r] - mx) * TAU_INV); sm += e[r]; }
    sm += __shfl_xor(sm, 16, 32);
    float inv = 1.0f / sm;

    // zero upper K-half of this lane's qt fragment row (16B store), then scatter qt
    *(uint4*)((char*)myQ + lane * 32 + 16) = uint4{0u, 0u, 0u, 0u};
    #pragma unroll
    for (int r = 0; r < 8; ++r) {
        float q = e[r] * inv;
        int laneF = (r + 8 * hi) + 16 * (col >> 3);
        myQ[laneF * 16 + (col & 7)] = (__bf16)q;
        float s = half_shfl_sum(q);
        if (col == 0) atomicAdd(&sH[r + 8 * hi], s);
    }
    __syncthreads();

    // mu numerator: D[m,dd] += qt (A, K zero-padded) x zt (B = this lane's regs!)
    const v16bf aq = *((const v16bf*)myQ + lane);
    #pragma unroll
    for (int t = 0; t < 8; ++t) {
        v16bf bz;
        #pragma unroll
        for (int i = 0; i < 16; ++i)
            bz[i] = (i < 8) ? (__bf16)acc[t][i] : (__bf16)0.0f;
        v8f an = {};
        an = __builtin_amdgcn_wmma_f32_16x16x32_bf16(false, aq, false, bz,
                                                     (short)0, an, false, false);
        #pragma unroll
        for (int r = 0; r < 8; ++r)
            atomicAdd(&sN[(r + 8 * hi) * 128 + t * 16 + col], an[r]);
    }
    __syncthreads();

    if (tid < 16) atomicAdd(&gHt[b * 16 + tid], sH[tid]);
    for (int i2 = tid; i2 < 2048; i2 += 256) atomicAdd(&gN[i2], sN[i2]);
}

// ---------------------------------------------------------------------------
// Finalize: denom, mu, C_new, KL
// ---------------------------------------------------------------------------
__global__ __launch_bounds__(256)
void finalize_kernel(const float* __restrict__ Cg, const float* __restrict__ gHt,
                     const float* __restrict__ gHs, const float* __restrict__ gN,
                     float* __restrict__ out)
{
    __shared__ float sden[16];
    __shared__ float sk[256];
    const int tid = threadIdx.x;

    if (tid < 16) {
        float d = 0.0f;
        for (int bb = 0; bb < 32; ++bb) d += gHt[bb * 16 + tid];
        sden[tid] = d + 1e-6f;
    }
    __syncthreads();

    if (tid < 16) {
        const int m = tid;
        const float den = sden[m];
        float ns = 0.0f;
        for (int dd = 0; dd < 128; ++dd) { float v = gN[m * 128 + dd] / den; ns += v * v; }
        const float invmu = 1.0f / fmaxf(sqrtf(ns), 1e-12f);
        float ns2 = 0.0f;
        for (int dd = 0; dd < 128; ++dd) {
            float mu = (gN[m * 128 + dd] / den) * invmu;
            float cn = 0.99f * Cg[m * 128 + dd] + 0.01f * mu;
            ns2 += cn * cn;
        }
        const float invc = 1.0f / fmaxf(sqrtf(ns2), 1e-12f);
        for (int dd = 0; dd < 128; ++dd) {
            float mu = (gN[m * 128 + dd] / den) * invmu;
            float cn = 0.99f * Cg[m * 128 + dd] + 0.01f * mu;
            out[1 + m * 128 + dd] = cn * invc;
        }
    }

    float part = 0.0f;
    for (int i = tid; i < 512; i += 256) {
        float ht = gHt[i] * (1.0f / 4096.0f);
        float hs = gHs[i] * (1.0f / 4096.0f);
        float t1 = (ht > 0.0f) ? ht * __logf(ht) : 0.0f;
        part += t1 - ht * __logf(hs + 1e-12f);
    }
    sk[tid] = part;
    __syncthreads();
    for (int s = 128; s > 0; s >>= 1) {
        if (tid < s) sk[tid] += sk[tid + s];
        __syncthreads();
    }
    if (tid == 0) out[0] = sk[0] * (1.0f / 32.0f);
}

extern "C" void kernel_launch(void* const* d_in, const int* in_sizes, int n_in,
                              void* d_out, int out_size, void* d_ws, size_t ws_size,
                              hipStream_t stream)
{
    (void)in_sizes; (void)n_in; (void)out_size; (void)ws_size;
    const float* fs4 = (const float*)d_in[0];
    const float* ft  = (const float*)d_in[1];
    const float* Ws  = (const float*)d_in[2];
    const float* Wt  = (const float*)d_in[3];
    const float* Cg  = (const float*)d_in[4];
    float* out = (float*)d_out;

    float* ws  = (float*)d_ws;
    float* gHt = ws;          // 512 floats
    float* gHs = ws + 512;    // 512 floats
    float* gN  = ws + 1024;   // 2048 floats

    hipMemsetAsync(d_ws, 0, 3072 * sizeof(float), stream);

    student_kernel<<<1024, 256, SMEM_S, stream>>>(fs4, Ws, Cg, gHs);
    teacher_kernel<<<1024, 256, SMEM_T, stream>>>(ft, Wt, Cg, gHt, gN);
    finalize_kernel<<<1, 256, 0, stream>>>(Cg, gHt, gHs, gN, out);
}
